// SparseConvNeXtBlock_17239998726329
// MI455X (gfx1250) — matrix operand order repaired and checked by reading, not compiled, not run on hardware
//
#include <hip/hip_runtime.h>
#include <hip/hip_bf16.h>
#include <stdint.h>

// ---------------------------------------------------------------------------
// SparseConvNeXt block, MI455X (gfx1250, wave32, WMMA).
// Sizes fixed by reference: B=2, C=192, D=H=W=32, K=7, HID=768.
// ---------------------------------------------------------------------------

typedef __attribute__((ext_vector_type(16))) _Float16 v16h;
typedef __attribute__((ext_vector_type(8)))  float    v8f;
typedef __attribute__((ext_vector_type(4)))  float    f4_t;

#define CC   192
#define HID  768
#define NTOK 65536           // B*D*H*W
#define NTILE16 (NTOK/16)    // 4096 token tiles of 16
#define PW1_HALVES (48*6*32*16)   // 147456
#define PW2_HALVES (12*24*32*16)  // 147456

// sched_group_barrier masks
#define SGB_MFMA      0x008
#define SGB_VMEM_READ 0x020

union V16U { f4_t f4[2]; v16h v; };

__device__ __forceinline__ void loadB(V16U& u, const _Float16* p) {
  u.f4[0] = ((const f4_t*)p)[0];
  u.f4[1] = ((const f4_t*)p)[1];
}

// Pipeline directive: VMEM4, (WMMA1, VMEM2)xN-1, WMMA1  (N fragments/wmmas)
template <int N>
__device__ __forceinline__ void sched_pipeline() {
  __builtin_amdgcn_sched_group_barrier(SGB_VMEM_READ, 4, 0);
  #pragma unroll
  for (int i = 0; i < N - 1; ++i) {
    __builtin_amdgcn_sched_group_barrier(SGB_MFMA, 1, 0);
    __builtin_amdgcn_sched_group_barrier(SGB_VMEM_READ, 2, 0);
  }
  __builtin_amdgcn_sched_group_barrier(SGB_MFMA, 1, 0);
}

// ---------------------------------------------------------------------------
// Kernel 0: pack w1 (192x768) and w2 (768x192) f32 -> f16 in the exact
// CDNA5 WMMA B-matrix (32x16, 16-bit) per-lane layout:
//   lanes 0-15 hold K {0..7,16..23}, lanes 16-31 hold K {8..15,24..31}.
// pw1 block (nt,kb): 32 lanes x 16 halves contiguous; same for pw2 (nb,k2b).
// ---------------------------------------------------------------------------
__global__ void pack_weights(const float* __restrict__ w1,
                             const float* __restrict__ w2,
                             _Float16* __restrict__ pw1,
                             _Float16* __restrict__ pw2) {
  int i = blockIdx.x * 256 + threadIdx.x;
  if (i < PW1_HALVES) {
    int j    = i & 15;
    int lane = (i >> 4) & 31;
    int blk  = i >> 9;            // nt*6 + kb
    int kb   = blk % 6;
    int nt   = blk / 6;
    int ck   = ((j < 8) ? j : (j + 8)) + ((lane >= 16) ? 8 : 0);
    int n    = lane & 15;
    pw1[i] = (_Float16)w1[(kb * 32 + ck) * HID + nt * 16 + n];
  } else if (i < PW1_HALVES + PW2_HALVES) {
    int i2   = i - PW1_HALVES;
    int j    = i2 & 15;
    int lane = (i2 >> 4) & 31;
    int blk  = i2 >> 9;           // nb*24 + k2b
    int k2b  = blk % 24;
    int nb   = blk / 24;
    int ck   = ((j < 8) ? j : (j + 8)) + ((lane >= 16) ? 8 : 0);
    int n    = lane & 15;
    pw2[i2] = (_Float16)w2[(k2b * 32 + ck) * CC + nb * 16 + n];
  }
}

// ---------------------------------------------------------------------------
// Kernel 1: depthwise 7^3 conv + bias + LayerNorm + mask, writing f16
// activations directly in the WMMA A-fragment layout (16x32 f16):
//   token r (0..15): lane r    holds K {0..7, 16..23} (halves j=0..15)
//                    lane r+16 holds K {8..15,24..31}
// One 64-thread block per (b, 4^3 mask cell). Inactive cells: zero-fill.
// ---------------------------------------------------------------------------
__global__ __launch_bounds__(64) void phase1_conv_ln(
    const float* __restrict__ x,
    const uint8_t* __restrict__ mask,      // numpy bool, 1 byte per cell
    const float* __restrict__ dw_w,        // (192, 343)
    const float* __restrict__ dw_b,
    const float* __restrict__ ln_w,
    const float* __restrict__ ln_b,
    _Float16* __restrict__ aBuf) {
  __shared__ float xt[1000];               // 10x10x10 padded input tile
  __shared__ float wt[343];
  __shared__ float yb[64 * 193];           // per-voxel conv results, padded

  const int blk  = blockIdx.x;             // 0..1023
  const int b    = blk >> 9;
  const int cell = blk & 511;
  const int cd = cell >> 6, ch = (cell >> 3) & 7, cw = cell & 7;
  const int tid = threadIdx.x;             // 0..63
  const int vd = tid >> 4, vh = (tid >> 2) & 3, vw = tid & 3;
  const int d = cd * 4 + vd, h = ch * 4 + vh, w = cw * 4 + vw;

  const int t      = ((b * 32 + d) * 32 + h) * 32 + w;
  const int tile16 = t >> 4;
  const int r      = t & 15;

  if (mask[blk] == 0) {
    // inactive cell: LN output is zeroed by the mask -> write zeros
    const f4_t z = {0.f, 0.f, 0.f, 0.f};
    #pragma unroll
    for (int kb = 0; kb < 6; ++kb) {
      _Float16* base = aBuf + (size_t)(tile16 * 6 + kb) * 512;  // 32 lanes * 16
      f4_t* p0 = (f4_t*)(base + r * 16);
      f4_t* p1 = (f4_t*)(base + (r + 16) * 16);
      p0[0] = z; p0[1] = z; p1[0] = z; p1[1] = z;
    }
    return;
  }

  const float* xb = x + (size_t)b * CC * 32768;
  float s1 = 0.f, s2 = 0.f;

  for (int c = 0; c < CC; ++c) {
    for (int i = tid; i < 343; i += 64) wt[i] = dw_w[c * 343 + i];
    const float* xc = xb + (size_t)c * 32768;
    for (int i = tid; i < 1000; i += 64) {
      int zd = i / 100, rem = i % 100, zh = rem / 10, zw = rem % 10;
      int gd = cd * 4 + zd - 3, gh = ch * 4 + zh - 3, gw = cw * 4 + zw - 3;
      float v = 0.f;
      if ((unsigned)gd < 32u && (unsigned)gh < 32u && (unsigned)gw < 32u)
        v = xc[(gd * 32 + gh) * 32 + gw];
      xt[i] = v;
    }
    __syncthreads();

    float acc = dw_b[c];
    for (int kd = 0; kd < 7; ++kd) {
      for (int kh = 0; kh < 7; ++kh) {
        const float* row = &xt[((vd + kd) * 10 + vh + kh) * 10 + vw];
        const float* wr  = &wt[(kd * 7 + kh) * 7];
        #pragma unroll
        for (int kw = 0; kw < 7; ++kw) acc = fmaf(row[kw], wr[kw], acc);
      }
    }
    yb[tid * 193 + c] = acc;
    s1 += acc;
    s2 += acc * acc;
    __syncthreads();
  }

  const float mu  = s1 * (1.f / 192.f);
  const float var = s2 * (1.f / 192.f) - mu * mu;
  const float rq  = rsqrtf(var + 1e-6f);

  // LN + pack into A-fragment layout, 4x16B stores per kb
  #pragma unroll
  for (int kb = 0; kb < 6; ++kb) {
    alignas(16) _Float16 hv[32];
    #pragma unroll
    for (int j = 0; j < 16; ++j) {
      int ck0 = (j < 8) ? j : (j + 8);    // lane r   : K {0..7,16..23}
      int ck1 = ck0 + 8;                  // lane r+16: K {8..15,24..31}
      int c0 = kb * 32 + ck0, c1 = kb * 32 + ck1;
      float l0 = (yb[tid * 193 + c0] - mu) * rq * ln_w[c0] + ln_b[c0];
      float l1 = (yb[tid * 193 + c1] - mu) * rq * ln_w[c1] + ln_b[c1];
      hv[j]      = (_Float16)l0;
      hv[16 + j] = (_Float16)l1;
    }
    _Float16* base = aBuf + (size_t)(tile16 * 6 + kb) * 512;
    f4_t* p0 = (f4_t*)(base + r * 16);
    f4_t* p1 = (f4_t*)(base + (r + 16) * 16);
    p0[0] = ((f4_t*)hv)[0]; p0[1] = ((f4_t*)hv)[1];
    p1[0] = ((f4_t*)hv)[2]; p1[1] = ((f4_t*)hv)[3];
  }
}

// ---------------------------------------------------------------------------
// Kernel 2: fused MLP via WMMA. 8 waves per WG, one 16-token tile per wave.
//   GEMM1: ln(16x192) @ w1(192x768), +b1, exact GELU
//   GEMM2: h(16x768)  @ w2(768x192), +b2, *gamma, *mask
// sched_group_barrier pipelines interleave B-fragment b128 loads with the
// WMMAs (VMEM4,(WMMA1,VMEM2)*,WMMA1) so fragments stay >=2 deep in flight
// instead of load->s_wait_loadcnt 0->wmma serialization. h is transposed
// D-layout -> A-layout through a per-wave LDS staging buffer (LDS is
// in-order within a wave; wave_barrier pins ordering and delimits the
// scheduling regions the pipeline directives apply to).
// ---------------------------------------------------------------------------
__global__ __launch_bounds__(256) void phase2_mlp(
    const _Float16* __restrict__ aBuf,
    const _Float16* __restrict__ pw1,
    const _Float16* __restrict__ pw2,
    const float* __restrict__ b1,
    const float* __restrict__ b2,
    const float* __restrict__ gamma,
    const uint8_t* __restrict__ mask,
    float* __restrict__ out) {
  __shared__ alignas(16) _Float16 ldsH[8][640];   // per-wave 1280 B staging

  const int tid  = threadIdx.x;
  const int wave = tid >> 5;
  const int lane = tid & 31;
  const int tile = blockIdx.x * 8 + wave;         // 0..4095
  const int t0   = tile * 16;
  const int b    = t0 >> 15;
  const int sp0  = t0 & 32767;

  // 4 consecutive mask cells cover this tile's 16 w-consecutive tokens
  const int d = sp0 >> 10, h = (sp0 >> 5) & 31, w0 = sp0 & 31;
  const int cellbase = b * 512 + (d >> 2) * 64 + (h >> 2) * 8 + (w0 >> 2);
  const unsigned mword = *(const unsigned*)(mask + cellbase);  // 4B aligned

  float* outb = out + (size_t)b * CC * 32768 + sp0;

  if (mword == 0) {  // wave-uniform: whole tile masked out -> zeros
    const f4_t z = {0.f, 0.f, 0.f, 0.f};
    #pragma unroll
    for (int nb = 0; nb < 12; ++nb) {
      float* p = outb + (size_t)(nb * 16 + (lane >> 1)) * 32768 + (lane & 1) * 8;
      ((f4_t*)p)[0] = z; ((f4_t*)p)[1] = z;
    }
    return;
  }

  const int laneoff = lane * 16;

  // warm WGP$ for the weight streams (global_prefetch_b8)
  __builtin_prefetch(pw1 + laneoff, 0, 1);
  __builtin_prefetch(pw2 + laneoff, 0, 1);

  // resident A1 fragments: 6 x v16h (K = 192)
  v16h afrag[6];
  #pragma unroll
  for (int kb = 0; kb < 6; ++kb) {
    const _Float16* p = aBuf + ((size_t)(tile * 6 + kb) * 32 + lane) * 16;
    V16U u; loadB(u, p);
    afrag[kb] = u.v;
  }

  v8f cout[12];
  #pragma unroll
  for (int nb = 0; nb < 12; ++nb) cout[nb] = (v8f){0,0,0,0,0,0,0,0};

  _Float16* hstage = ldsH[wave];                  // 16 rows x 40 halves
  const int ncol    = lane & 15;
  const int mrow_lo = (lane < 16) ? 0 : 8;        // D-layout M offset
  const int koff    = (lane < 16) ? 0 : 8;        // A-layout K offset

  for (int ntp = 0; ntp < 24; ++ntp) {            // pairs of hid 16-col tiles
    // ---- region 1: GEMM1 (2 x 6 WMMA, 2 x 12 b128 reads) + GELU + ds writes
    #pragma unroll
    for (int half = 0; half < 2; ++half) {
      const int nt = ntp * 2 + half;
      const _Float16* pB1 = pw1 + (size_t)nt * 6 * 512 + laneoff;

      v8f hc = (v8f){0,0,0,0,0,0,0,0};
      V16U bb[2];
      loadB(bb[0], pB1);
      #pragma unroll
      for (int kb = 0; kb < 6; ++kb) {
        if (kb < 5) loadB(bb[(kb + 1) & 1], pB1 + (kb + 1) * 512);
        hc = __builtin_amdgcn_wmma_f32_16x16x32_f16(
            false, afrag[kb], false, bb[kb & 1].v, (short)0, hc, false, false);
      }
      const float bias = b1[nt * 16 + ncol];
      #pragma unroll
      for (int i = 0; i < 8; ++i) {
        float v = hc[i] + bias;
        v = 0.5f * v * (1.0f + erff(v * 0.70710678118654752f));  // exact GELU
        hstage[(i + mrow_lo) * 40 + half * 16 + ncol] = (_Float16)v;
      }
    }
    sched_pipeline<12>();                         // pipeline both halves
    __builtin_amdgcn_wave_barrier();              // pin store->load ordering

    // ---- region 2: reload h tile as A-fragment (LDS transpose, per wave)
    const int m = lane & 15;
    V16U a2;
    a2.f4[0] = *(const f4_t*)&hstage[m * 40 + koff];        // K koff..koff+7
    a2.f4[1] = *(const f4_t*)&hstage[m * 40 + 16 + koff];   // K 16+koff..
    __builtin_amdgcn_wave_barrier();

    // ---- region 3: GEMM2 (12 WMMA, 24 b128 reads)
    const _Float16* pB2 = pw2 + (size_t)ntp * 512 + laneoff;
    V16U cb[2];
    loadB(cb[0], pB2);
    #pragma unroll
    for (int nb = 0; nb < 12; ++nb) {
      if (nb < 11) loadB(cb[(nb + 1) & 1], pB2 + (size_t)(nb + 1) * 24 * 512);
      cout[nb] = __builtin_amdgcn_wmma_f32_16x16x32_f16(
          false, a2.v, false, cb[nb & 1].v, (short)0, cout[nb], false, false);
    }
    sched_pipeline<12>();
  }

  // epilogue: +b2, *gamma, per-token mask, LDS transpose, coalesced store
  float* ostage = (float*)ldsH[wave];             // 16 rows x 20 floats
  #pragma unroll
  for (int nb = 0; nb < 12; ++nb) {
    const int c  = nb * 16 + ncol;
    const float bb = b2[c], gg = gamma[c];
    #pragma unroll
    for (int i = 0; i < 8; ++i) {
      const int M = i + mrow_lo;                  // token index in tile
      float v = gg * (cout[nb][i] + bb);
      const unsigned mb = (mword >> ((M >> 2) * 8)) & 0xffu;
      v = mb ? v : 0.f;
      ostage[ncol * 20 + M] = v;                  // [channel][token]
    }
    __builtin_amdgcn_wave_barrier();
    const int crow = lane >> 1;
    const int toff = (lane & 1) * 8;
    const f4_t v0 = ((const f4_t*)&ostage[crow * 20 + toff])[0];
    const f4_t v1 = ((const f4_t*)&ostage[crow * 20 + toff])[1];
    float* p = outb + (size_t)(nb * 16 + crow) * 32768 + toff;
    ((f4_t*)p)[0] = v0; ((f4_t*)p)[1] = v1;
    __builtin_amdgcn_wave_barrier();
  }
}

// ---------------------------------------------------------------------------
// Launch. Input order (setup_inputs): x, mask, dw_w, dw_b, ln_w, ln_b,
//                                     w1, b1, w2, b2, gamma
// Workspace: [aBuf f16 25.17MB][pw1 288KB][pw2 288KB] ~= 25.8 MB
// ---------------------------------------------------------------------------
extern "C" void kernel_launch(void* const* d_in, const int* in_sizes, int n_in,
                              void* d_out, int out_size, void* d_ws, size_t ws_size,
                              hipStream_t stream) {
  const float*   x    = (const float*)d_in[0];
  const uint8_t* mask = (const uint8_t*)d_in[1];   // numpy bool, 1B/elem
  const float*   dw_w = (const float*)d_in[2];
  const float*   dw_b = (const float*)d_in[3];
  const float*   ln_w = (const float*)d_in[4];
  const float*   ln_b = (const float*)d_in[5];
  const float*   w1   = (const float*)d_in[6];
  const float*   b1   = (const float*)d_in[7];
  const float*   w2   = (const float*)d_in[8];
  const float*   b2   = (const float*)d_in[9];
  const float*   gamma= (const float*)d_in[10];
  float* out = (float*)d_out;

  uint8_t* ws = (uint8_t*)d_ws;
  _Float16* aBuf = (_Float16*)ws;                                  // 25165824 B
  _Float16* pw1  = (_Float16*)(ws + (((size_t)NTOK * CC * 2 + 255) & ~(size_t)255));
  _Float16* pw2  = pw1 + PW1_HALVES;

  pack_weights<<<(PW1_HALVES + PW2_HALVES + 255) / 256, 256, 0, stream>>>(
      w1, w2, pw1, pw2);
  phase1_conv_ln<<<2 * 512, 64, 0, stream>>>(
      x, mask, dw_w, dw_b, ln_w, ln_b, aBuf);
  phase2_mlp<<<NTILE16 / 8, 256, 0, stream>>>(
      aBuf, pw1, pw2, b1, b2, gamma, mask, out);
}
